// GOTSim_87694642250043
// MI455X (gfx1250) — matrix operand order repaired and checked by reading, not compiled, not run on hardware
//
#include <hip/hip_runtime.h>
#include <math.h>

// ---------------------------------------------------------------------------
// Types for CDNA5 WMMA (wave32, 16x16x32 f16 -> f32)
// ---------------------------------------------------------------------------
typedef __attribute__((ext_vector_type(16))) _Float16 v16h;
typedef __attribute__((ext_vector_type(8)))  float    v8f;

#define GB   128          // batch of graph pairs
#define NL   3            // GNN layers
#define FDIM 512          // hidden width
#define BIGC 99999.0f
#define HBIG 1e30f

__device__ __forceinline__ v8f wmma16(v16h a, v16h b, v8f c) {
    // 8 args: (neg_a, A, neg_b, B, c_mod, C, reuse_a, reuse_b)
    return __builtin_amdgcn_wmma_f32_16x16x32_f16(
        false, a, false, b, (short)0, c, false, false);
}

// split fp32 -> (hi, lo) fp16 pair; hi+lo ~= x with ~22 bits of mantissa
struct h2 { _Float16 hi, lo; };
__device__ __forceinline__ h2 hsplit(float x) {
    _Float16 h = (_Float16)x;
    h2 r;
    r.hi = h;
    r.lo = (_Float16)(x - (float)h);
    return r;
}

// ---------------------------------------------------------------------------
// Prefix sums of graph sizes (tiny, serial)
// ---------------------------------------------------------------------------
__global__ void k_prefix(const int* __restrict__ qs, const int* __restrict__ cs,
                         int* __restrict__ qoffs, int* __restrict__ coffs) {
    if (blockIdx.x == 0 && threadIdx.x == 0) {
        int a = 0, b = 0;
        for (int i = 0; i < GB; ++i) {
            qoffs[i] = a; a += qs[i];
            coffs[i] = b; b += cs[i];
        }
        qoffs[GB] = a; coffs[GB] = b;
    }
}

// ---------------------------------------------------------------------------
// Degree / dinv
// ---------------------------------------------------------------------------
__global__ void k_fill1(float* __restrict__ deg, int n) {
    int i = blockIdx.x * blockDim.x + threadIdx.x;
    if (i < n) deg[i] = 1.0f;
}
__global__ void k_count(float* __restrict__ deg, const int* __restrict__ dst, int e) {
    int i = blockIdx.x * blockDim.x + threadIdx.x;
    if (i < e) atomicAdd(&deg[dst[i]], 1.0f);
}
__global__ void k_rsqrt(float* __restrict__ d, int n) {
    int i = blockIdx.x * blockDim.x + threadIdx.x;
    if (i < n) d[i] = rsqrtf(d[i]);
}

// ---------------------------------------------------------------------------
// WMMA GEMM: Out(N x 512) = X(N x K) @ W(K x 512), K in {256,512}
// One wave per 16x16 output tile, fp16 hi/lo split (3 WMMA per K-step).
// ---------------------------------------------------------------------------
__global__ void k_gemm(const float* __restrict__ X, int K,
                       const float* __restrict__ W,
                       float* __restrict__ Out, int N) {
    const int lane = threadIdx.x & 31;
    const int r0 = blockIdx.x * 16;
    const int c0 = blockIdx.y * 16;

    int row = r0 + (lane & 15);
    if (row >= N) row = N - 1;                 // clamp; store is guarded
    const float* xrow = X + (size_t)row * K;
    const int kb = (lane >> 4) << 3;           // 0 or 8

    v8f acc = {};
    for (int k0 = 0; k0 < K; k0 += 32) {
        __builtin_prefetch(xrow + k0 + 64, 0, 1);   // global_prefetch_b8
        v16h ah, al, bh, bl;
#pragma unroll
        for (int e = 0; e < 8; ++e) {
            h2 t0 = hsplit(xrow[k0 + kb + e]);
            ah[e] = t0.hi;     al[e] = t0.lo;
            h2 t1 = hsplit(xrow[k0 + kb + 16 + e]);
            ah[8 + e] = t1.hi; al[8 + e] = t1.lo;
        }
        const float* wrow = W + (size_t)k0 * FDIM + c0 + (lane & 15);
#pragma unroll
        for (int e = 0; e < 8; ++e) {
            h2 t0 = hsplit(wrow[(size_t)(kb + e) * FDIM]);
            bh[e] = t0.hi;     bl[e] = t0.lo;
            h2 t1 = hsplit(wrow[(size_t)(kb + 16 + e) * FDIM]);
            bh[8 + e] = t1.hi; bl[8 + e] = t1.lo;
        }
        acc = wmma16(ah, bh, acc);
        acc = wmma16(ah, bl, acc);
        acc = wmma16(al, bh, acc);
    }
    const int rb = r0 + ((lane >> 4) << 3);
    const int n  = c0 + (lane & 15);
#pragma unroll
    for (int i = 0; i < 8; ++i) {
        int r = rb + i;
        if (r < N) Out[(size_t)r * FDIM + n] = acc[i];
    }
}

// ---------------------------------------------------------------------------
// GCN self term: raw = tmp * (1/deg) + bias      (1/deg = dinv^2)
// ---------------------------------------------------------------------------
__global__ void k_self(const float* __restrict__ tmp, const float* __restrict__ dinv,
                       const float* __restrict__ bias, float* __restrict__ out, int N) {
    size_t idx = (size_t)blockIdx.x * blockDim.x + threadIdx.x;
    if (idx >= (size_t)N * FDIM) return;
    int i = (int)(idx >> 9);
    int f = (int)(idx & (FDIM - 1));
    float di = dinv[i];
    out[idx] = tmp[idx] * di * di + bias[f];
}

// GCN message scatter: out[dst] += tmp[src] * dinv[src]*dinv[dst]
__global__ void k_scatter(const float* __restrict__ tmp, float* __restrict__ out,
                          const int* __restrict__ src, const int* __restrict__ dst,
                          const float* __restrict__ dinv, int E) {
    int e = blockIdx.x;
    if (e >= E) return;
    int s = src[e], d = dst[e];
    float w = dinv[s] * dinv[d];
    const float* hp = tmp + (size_t)s * FDIM;
    float*       op = out + (size_t)d * FDIM;
    for (int f = threadIdx.x; f < FDIM; f += blockDim.x)
        atomicAdd(&op[f], hp[f] * w);
}

__global__ void k_relu(const float* __restrict__ in, float* __restrict__ out, size_t n) {
    size_t i = (size_t)blockIdx.x * blockDim.x + threadIdx.x;
    if (i < n) { float x = in[i]; out[i] = x > 0.f ? x : 0.f; }
}

// ---------------------------------------------------------------------------
// Cost tensors for one layer:
//   A[l,b,n,m]  = -dot(Qraw[b,n,:], Craw[b,m,:])  (WMMA, 4 waves = 2x2 tiles)
//   Bd[l,b,n]   = -dot(Qraw[b,n,:], del_p[l,:])
//   Cd[l,b,m]   = -dot(Craw[b,m,:], ins_p[l,:])
// ---------------------------------------------------------------------------
__global__ void k_cost(const float* __restrict__ Qh, const float* __restrict__ Ch,
                       const int* __restrict__ qoffs, const int* __restrict__ coffs,
                       const int* __restrict__ qs, const int* __restrict__ cs,
                       const float* __restrict__ delp, const float* __restrict__ insp,
                       float* __restrict__ A, float* __restrict__ Bd,
                       float* __restrict__ Cd, int l) {
    const int b  = blockIdx.x;
    const int s1 = qs[b], s2 = cs[b];
    const int qo = qoffs[b], co = coffs[b];
    const int tid  = threadIdx.x;
    const int wave = tid >> 5;
    const int lane = tid & 31;
    const int n0 = (wave >> 1) * 16;
    const int m0 = (wave & 1) * 16;

    int rn = n0 + (lane & 15);
    if (rn >= s1) rn = s1 - 1;                 // clamp (invalid region never read)
    const float* qrow = Qh + (size_t)(qo + rn) * FDIM;
    const int kb = (lane >> 4) << 3;

    v8f acc = {};
    for (int k0 = 0; k0 < FDIM; k0 += 32) {
        v16h ah, al, bh, bl;
#pragma unroll
        for (int e = 0; e < 8; ++e) {
            h2 t0 = hsplit(qrow[k0 + kb + e]);
            ah[e] = t0.hi;     al[e] = t0.lo;
            h2 t1 = hsplit(qrow[k0 + kb + 16 + e]);
            ah[8 + e] = t1.hi; al[8 + e] = t1.lo;
        }
        int mc = m0 + (lane & 15);
        if (mc >= s2) mc = s2 - 1;
        const float* crow = Ch + (size_t)(co + mc) * FDIM + k0 + kb;
#pragma unroll
        for (int e = 0; e < 8; ++e) {
            h2 t0 = hsplit(crow[e]);
            bh[e] = t0.hi;     bl[e] = t0.lo;
            h2 t1 = hsplit(crow[16 + e]);
            bh[8 + e] = t1.hi; bl[8 + e] = t1.lo;
        }
        acc = wmma16(ah, bh, acc);
        acc = wmma16(ah, bl, acc);
        acc = wmma16(al, bh, acc);
    }
    float* Ab = A + ((size_t)(l * GB + b) << 10);  // 32*32 per (l,b)
    const int nb = n0 + ((lane >> 4) << 3);
    const int m  = m0 + (lane & 15);
#pragma unroll
    for (int i = 0; i < 8; ++i) Ab[(nb + i) * 32 + m] = -acc[i];

    // diagonal delete / insert costs
    if (tid < 32) {
        int n = tid, r = (n < s1) ? n : s1 - 1;
        const float* qr = Qh + (size_t)(qo + r) * FDIM;
        float s = 0.f;
        for (int d = 0; d < FDIM; ++d) s += qr[d] * delp[d];
        Bd[(size_t)(l * GB + b) * 32 + n] = -s;
    } else if (tid < 64) {
        int mm = tid - 32, r = (mm < s2) ? mm : s2 - 1;
        const float* cr = Ch + (size_t)(co + r) * FDIM;
        float s = 0.f;
        for (int d = 0; d < FDIM; ++d) s += cr[d] * insp[d];
        Cd[(size_t)(l * GB + b) * 32 + mm] = -s;
    }
}

// ---------------------------------------------------------------------------
// Hungarian (shortest augmenting path), one wave32 per (l,b) problem.
// Cost matrix (<=65x65) in LDS; column ops parallel over lanes; argmin via
// __shfl_xor. Single-wave workgroup => __syncthreads lowers to S_NOP.
// Emits mcost[l,b] = total assignment cost == sum of the reference einsums.
// ---------------------------------------------------------------------------
__global__ void k_hung(const float* __restrict__ A, const float* __restrict__ Bd,
                       const float* __restrict__ Cd, const int* __restrict__ qs,
                       const int* __restrict__ cs, float* __restrict__ mcost) {
    const int bid = blockIdx.x;
    const int b  = bid & (GB - 1);
    const int s1 = qs[b], s2 = cs[b];
    const int nn = s1 + s2;                    // <= 64
    const int lane = threadIdx.x;

    __shared__ float Cmat[65 * 65];
    __shared__ float u[66], v[66], minv[66];
    __shared__ int   way[66], p[66], used[66];

    const float* Ab = A  + ((size_t)bid << 10);
    const float* Bb = Bd + (size_t)bid * 32;
    const float* Cb = Cd + (size_t)bid * 32;

    const int tot = (nn + 1) * (nn + 1);
    for (int t = lane; t < tot; t += 32) {
        int i = t / (nn + 1), j = t - i * (nn + 1);
        float val = 0.f;
        if (i >= 1 && j >= 1) {
            int r = i - 1, c = j - 1;
            if (r < s1) {
                if (c < s2) val = Ab[r * 32 + c];
                else        val = ((c - s2) == r) ? Bb[r] : BIGC;
            } else {
                int r2 = r - s1;
                if (c < s2) val = (r2 == c) ? Cb[r2] : BIGC;
                // else bottom-right block: 0
            }
        }
        Cmat[i * 65 + j] = val;
    }
    for (int t = lane; t <= nn; t += 32) { u[t] = 0.f; v[t] = 0.f; p[t] = 0; }
    __syncthreads();

    for (int i = 1; i <= nn; ++i) {
        if (lane == 0) p[0] = i;
        for (int t = lane; t <= nn; t += 32) { minv[t] = HBIG; way[t] = 0; used[t] = 0; }
        __syncthreads();
        int j0 = 0;
        while (true) {
            if (lane == 0) used[j0] = 1;
            __syncthreads();
            const int i0 = p[j0];
            const float ui0 = u[i0];
            for (int j = lane; j <= nn; j += 32) {
                if (!used[j]) {
                    float cur = Cmat[i0 * 65 + j] - ui0 - v[j];
                    if (cur < minv[j]) { minv[j] = cur; way[j] = j0; }
                }
            }
            __syncthreads();
            // argmin over unused columns j>=1, tie -> smallest j (matches np.argmin)
            float bv = HBIG; int bj = 0x7fffffff;
            for (int j = lane; j <= nn; j += 32) {
                if (j >= 1 && !used[j]) {
                    float x = minv[j];
                    if (x < bv || (x == bv && j < bj)) { bv = x; bj = j; }
                }
            }
            for (int off = 16; off; off >>= 1) {
                float ov = __shfl_xor(bv, off, 32);
                int   oj = __shfl_xor(bj, off, 32);
                if (ov < bv || (ov == bv && oj < bj)) { bv = ov; bj = oj; }
            }
            j0 = bj;
            const float delta = bv;
            for (int j = lane; j <= nn; j += 32) {
                if (used[j]) { u[p[j]] += delta; v[j] -= delta; }  // p[j] distinct
                else         minv[j] -= delta;
            }
            __syncthreads();
            if (p[j0] == 0) break;
        }
        if (lane == 0) {                       // augment along 'way'
            int jj = j0;
            while (jj != 0) { int j1 = way[jj]; p[jj] = p[j1]; jj = j1; }
        }
        __syncthreads();
    }
    float s = 0.f;
    for (int j = lane; j <= nn; j += 32)
        if (j >= 1) s += Cmat[p[j] * 65 + j];
    for (int off = 16; off; off >>= 1) s += __shfl_xor(s, off, 32);
    if (lane == 0) mcost[bid] = s;
}

// ---------------------------------------------------------------------------
// Final scores: sigmoid(sum_l ot_w[l] * 2*mcost[l,b]/(s1+s2) + ot_b)
// ---------------------------------------------------------------------------
__global__ void k_score(const float* __restrict__ mcost, const int* __restrict__ qs,
                        const int* __restrict__ cs, const float* __restrict__ otw,
                        const float* __restrict__ otb, float* __restrict__ out) {
    int b = blockIdx.x * blockDim.x + threadIdx.x;
    if (b >= GB) return;
    float sz = (float)(qs[b] + cs[b]);
    float s = otb[0];
    for (int l = 0; l < NL; ++l)
        s += otw[l] * (2.0f * mcost[l * GB + b] / sz);
    out[b] = 1.0f / (1.0f + __expf(-s));
}

// ---------------------------------------------------------------------------
// Host orchestration
// ---------------------------------------------------------------------------
extern "C" void kernel_launch(void* const* d_in, const int* in_sizes, int n_in,
                              void* d_out, int out_size, void* d_ws, size_t ws_size,
                              hipStream_t stream) {
    (void)n_in; (void)out_size; (void)ws_size;

    const float* qx     = (const float*)d_in[0];
    const float* cx     = (const float*)d_in[1];
    const int*   q_src  = (const int*)d_in[2];
    const int*   q_dst  = (const int*)d_in[3];
    const int*   c_src  = (const int*)d_in[6];
    const int*   c_dst  = (const int*)d_in[7];
    const int*   qsizes = (const int*)d_in[10];
    const int*   csizes = (const int*)d_in[11];
    const float* Ws[NL] = {(const float*)d_in[12], (const float*)d_in[14], (const float*)d_in[16]};
    const float* bs[NL] = {(const float*)d_in[13], (const float*)d_in[15], (const float*)d_in[17]};
    const float* ins_p  = (const float*)d_in[18];
    const float* del_p  = (const float*)d_in[19];
    const float* ot_w   = (const float*)d_in[20];
    const float* ot_b   = (const float*)d_in[21];

    const int Nq = in_sizes[0] / 256;
    const int Nc = in_sizes[1] / 256;
    const int Eq = in_sizes[2];
    const int Ec = in_sizes[6];

    // Carve workspace (256B-aligned slices)
    char* wp = (char*)d_ws;
    auto carve = [&](size_t bytes) -> void* {
        void* r = (void*)wp;
        wp += (bytes + 255) & ~(size_t)255;
        return r;
    };
    int*   qoffs  = (int*)  carve((GB + 1) * sizeof(int));
    int*   coffs  = (int*)  carve((GB + 1) * sizeof(int));
    float* dinv_q = (float*)carve((size_t)Nq * sizeof(float));
    float* dinv_c = (float*)carve((size_t)Nc * sizeof(float));
    float* hq_act = (float*)carve((size_t)Nq * FDIM * sizeof(float));
    float* hq_tmp = (float*)carve((size_t)Nq * FDIM * sizeof(float));
    float* hq_raw = (float*)carve((size_t)Nq * FDIM * sizeof(float));
    float* hc_act = (float*)carve((size_t)Nc * FDIM * sizeof(float));
    float* hc_tmp = (float*)carve((size_t)Nc * FDIM * sizeof(float));
    float* hc_raw = (float*)carve((size_t)Nc * FDIM * sizeof(float));
    float* Amat   = (float*)carve((size_t)NL * GB * 32 * 32 * sizeof(float));
    float* Bdia   = (float*)carve((size_t)NL * GB * 32 * sizeof(float));
    float* Cdia   = (float*)carve((size_t)NL * GB * 32 * sizeof(float));
    float* mcost  = (float*)carve((size_t)NL * GB * sizeof(float));

    // Graph offsets + degrees (dinv = rsqrt(1 + in-degree))
    k_prefix<<<1, 1, 0, stream>>>(qsizes, csizes, qoffs, coffs);
    k_fill1<<<(Nq + 255) / 256, 256, 0, stream>>>(dinv_q, Nq);
    k_fill1<<<(Nc + 255) / 256, 256, 0, stream>>>(dinv_c, Nc);
    k_count<<<(Eq + 255) / 256, 256, 0, stream>>>(dinv_q, q_dst, Eq);
    k_count<<<(Ec + 255) / 256, 256, 0, stream>>>(dinv_c, c_dst, Ec);
    k_rsqrt<<<(Nq + 255) / 256, 256, 0, stream>>>(dinv_q, Nq);
    k_rsqrt<<<(Nc + 255) / 256, 256, 0, stream>>>(dinv_c, Nc);

    for (int l = 0; l < NL; ++l) {
        const int Fin = (l == 0) ? 256 : FDIM;
        const float* aq = (l == 0) ? qx : hq_act;
        const float* ac = (l == 0) ? cx : hc_act;

        // h = act @ W  (WMMA)
        k_gemm<<<dim3((Nq + 15) / 16, FDIM / 16), 32, 0, stream>>>(aq, Fin, Ws[l], hq_tmp, Nq);
        k_gemm<<<dim3((Nc + 15) / 16, FDIM / 16), 32, 0, stream>>>(ac, Fin, Ws[l], hc_tmp, Nc);

        // raw = scatter(msg) + h/deg + bias
        k_self<<<(int)(((size_t)Nq * FDIM + 255) / 256), 256, 0, stream>>>(hq_tmp, dinv_q, bs[l], hq_raw, Nq);
        k_self<<<(int)(((size_t)Nc * FDIM + 255) / 256), 256, 0, stream>>>(hc_tmp, dinv_c, bs[l], hc_raw, Nc);
        k_scatter<<<Eq, 256, 0, stream>>>(hq_tmp, hq_raw, q_src, q_dst, dinv_q, Eq);
        k_scatter<<<Ec, 256, 0, stream>>>(hc_tmp, hc_raw, c_src, c_dst, dinv_c, Ec);

        // cost tensors for this layer (pre-relu features, per reference)
        k_cost<<<GB, 128, 0, stream>>>(hq_raw, hc_raw, qoffs, coffs, qsizes, csizes,
                                       del_p + (size_t)l * FDIM, ins_p + (size_t)l * FDIM,
                                       Amat, Bdia, Cdia, l);

        if (l < NL - 1) {
            k_relu<<<(int)(((size_t)Nq * FDIM + 255) / 256), 256, 0, stream>>>(hq_raw, hq_act, (size_t)Nq * FDIM);
            k_relu<<<(int)(((size_t)Nc * FDIM + 255) / 256), 256, 0, stream>>>(hc_raw, hc_act, (size_t)Nc * FDIM);
        }
    }

    // 384 independent Hungarian problems, one wave32 each
    k_hung<<<NL * GB, 32, 0, stream>>>(Amat, Bdia, Cdia, qsizes, csizes, mcost);

    // Final 128 scores
    k_score<<<1, GB, 0, stream>>>(mcost, qsizes, csizes, ot_w, ot_b, (float*)d_out);
}
